// MultiHeadAttention_78589311582490
// MI455X (gfx1250) — matrix operand order repaired and checked by reading, not compiled, not run on hardware
//
#include <hip/hip_runtime.h>
#include <cmath>

// ---------------------------------------------------------------------------
// MI455X (gfx1250) multi-head attention, space-time block-causal mask.
// All matrix math on V_WMMA_F32_16X16X32_BF16 (bf16 in, f32 accumulate).
// bf16 operands pre-converted once into d_ws; V kept transposed per head so
// every WMMA fragment is built from 16B contiguous loads.
// Outputs (flat, in order): qh, kh, vh, attention, out.
// ---------------------------------------------------------------------------

typedef __attribute__((ext_vector_type(16))) __bf16 v16bf;
typedef __attribute__((ext_vector_type(8)))  __bf16 v8bf;
typedef __attribute__((ext_vector_type(8)))  float  v8f;

#define HN  12
#define DKN 64
#define DN  768
#define NPN 196
#define SN  3136            // 16 frames * 196 patches
#define SD  (SN * DN)       // 2,408,448
#define WSZ (DN * DN)       // 589,824

__device__ __forceinline__ v16bf cat16(v8bf a, v8bf b) {
  return __builtin_shufflevector(a, b, 0, 1, 2, 3, 4, 5, 6, 7,
                                       8, 9, 10, 11, 12, 13, 14, 15);
}

__device__ __forceinline__ v8f wmma_bf16(v16bf a, v16bf b, v8f c) {
  return __builtin_amdgcn_wmma_f32_16x16x32_bf16(false, a, false, b,
                                                 (short)0, c, false, false);
}

// ---------------------------------------------------------------------------
// Bulk f32 -> bf16 conversion (n divisible by 8; vectorized 8 / thread).
// ---------------------------------------------------------------------------
__global__ void __launch_bounds__(256)
cvt_f32_bf16(const float* __restrict__ in, __bf16* __restrict__ out, int n)
{
  const int i = (blockIdx.x * 256 + threadIdx.x) * 8;
  if (i >= n) return;
  float4 x = *(const float4*)(in + i);
  float4 y = *(const float4*)(in + i + 4);
  v8bf o;
  o[0] = (__bf16)x.x; o[1] = (__bf16)x.y; o[2] = (__bf16)x.z; o[3] = (__bf16)x.w;
  o[4] = (__bf16)y.x; o[5] = (__bf16)y.y; o[6] = (__bf16)y.z; o[7] = (__bf16)y.w;
  *(v8bf*)(out + i) = o;
}

// ---------------------------------------------------------------------------
// GEMM: out = X[SN,DN](bf16) @ W[DN,DN](bf16)^T + bias. One wave / 16x16 tile.
// MODE 0: f32 head layout [H][SN][64] + bf16 copy (same layout)      (Q, K)
// MODE 2: f32 head layout [H][SN][64] + bf16 TRANSPOSED [H][64][SN]  (V)
// MODE 1: f32 row-major [SN,DN] only                                 (out)
// ---------------------------------------------------------------------------
template <int MODE>
__global__ void __launch_bounds__(32)
gemm_bf16(const __bf16* __restrict__ X, const __bf16* __restrict__ W,
          const float* __restrict__ bias, float* __restrict__ out,
          __bf16* __restrict__ outb)
{
  const int lane = threadIdx.x & 31;
  const int ln = lane & 15;          // A-row / B-col within tile
  const int g  = lane >> 4;          // half-wave group
  const int m0 = blockIdx.x * 16;
  const int n0 = blockIdx.y * 16;

  const __bf16* ap = X + (m0 + ln) * DN + 16 * g;   // A: K = 16g+[0,16)
  const __bf16* bp = W + (n0 + ln) * DN + 16 * g;   // B: K = 16g+[0,16)

  v8f c = {};
  #pragma unroll 4
  for (int kk = 0; kk < DN; kk += 32) {
    v16bf a = cat16(*(const v8bf*)(ap + kk), *(const v8bf*)(ap + kk + 8));
    v16bf b = cat16(*(const v8bf*)(bp + kk), *(const v8bf*)(bp + kk + 8));
    c = wmma_bf16(a, b, c);
  }

  const float bv = bias[n0 + ln];
  const int col = n0 + ln;
  #pragma unroll
  for (int r = 0; r < 8; ++r) {
    const int row = m0 + 8 * g + r;
    const float val = c[r] + bv;
    if constexpr (MODE == 1) {
      out[row * DN + col] = val;
    } else {
      const int idx = (col >> 6) * (SN * 64) + row * 64 + (col & 63);
      out[idx] = val;
      if constexpr (MODE == 0) outb[idx] = (__bf16)val;
    }
  }
  if constexpr (MODE == 2) {
    // transposed bf16 copy: rows m0+8g+[0,8) are contiguous -> one 16B store
    v8bf tv;
    #pragma unroll
    for (int r = 0; r < 8; ++r) tv[r] = (__bf16)(c[r] + bv);
    *(v8bf*)&outb[((col >> 6) * 64 + (col & 63)) * SN + m0 + 8 * g] = tv;
  }
}

// ---------------------------------------------------------------------------
// Attention: one wave per (head, 16-query-row block).
// Phase 1: branchless per-lane online softmax over allowed key tiles.
// Phase 2: recompute scores, write normalized P (f32) to d_out, stage P in
//          LDS as bf16, reload in WMMA-A layout, accumulate O += P @ V.
// ---------------------------------------------------------------------------
__global__ void __launch_bounds__(32)
attn_kernel(const __bf16* __restrict__ qhb, const __bf16* __restrict__ khb,
            const __bf16* __restrict__ vtb, float* __restrict__ attn,
            __bf16* __restrict__ oconb)
{
  __shared__ __bf16 P[16][40];       // 16x32 P tile, padded row stride

  const int lane = threadIdx.x & 31;
  const int ln = lane & 15;
  const int g  = lane >> 4;
  const int h  = blockIdx.y;
  const int m0 = blockIdx.x * 16;

  const __bf16* q  = qhb + h * (SN * DKN);
  const __bf16* k  = khb + h * (SN * DKN);
  const __bf16* vt = vtb + h * (DKN * SN);   // [64][SN] transposed
  float*        at = attn + (size_t)h * SN * SN;

  // Q fragments (fixed for the whole row block): kk = 0 and kk = 32
  v16bf qa[2];
  #pragma unroll
  for (int i = 0; i < 2; ++i) {
    const __bf16* ap = q + (m0 + ln) * DKN + i * 32 + 16 * g;
    qa[i] = cat16(*(const v8bf*)ap, *(const v8bf*)(ap + 8));
  }

  int qf[8];
  #pragma unroll
  for (int r = 0; r < 8; ++r) qf[r] = (m0 + 8 * g + r) / NPN;

  const int maxF     = (m0 + 15) / NPN;
  const int colLimit = (maxF + 1) * NPN;       // keys beyond this are masked
  const int nT       = (colLimit + 31) / 32;   // 32-key column tiles

  // ---------------- phase 1: branchless per-lane online softmax -----------
  // Safe: tile 0 (frame 0) is always allowed, so ml becomes finite before
  // any masked (-inf) score is folded in -> no (-inf) - (-inf).
  float ml[8], sl[8];
  #pragma unroll
  for (int r = 0; r < 8; ++r) { ml[r] = -INFINITY; sl[r] = 0.f; }

  for (int t = 0; t < nT; ++t) {
    const int c0 = t * 32;
    #pragma unroll
    for (int b = 0; b < 2; ++b) {
      const int col = c0 + 16 * b + ln;
      const int kf  = col / NPN;
      const __bf16* bp = k + col * DKN + 16 * g;
      v8f s = {};
      s = wmma_bf16(qa[0], cat16(*(const v8bf*)bp, *(const v8bf*)(bp + 8)), s);
      s = wmma_bf16(qa[1], cat16(*(const v8bf*)(bp + 32), *(const v8bf*)(bp + 40)), s);
      #pragma unroll
      for (int r = 0; r < 8; ++r) {
        const float sv = (kf <= qf[r]) ? s[r] * 0.125f : -INFINITY;
        const float mn = fmaxf(ml[r], sv);
        sl[r] = sl[r] * __expf(ml[r] - mn) + __expf(sv - mn);
        ml[r] = mn;
      }
    }
  }

  // merge (max,sum) across the 16 lanes sharing each row
  float rmax[8], rinv[8];
  #pragma unroll
  for (int r = 0; r < 8; ++r) {
    float om = ml[r], os = sl[r];
    #pragma unroll
    for (int msk = 1; msk < 16; msk <<= 1) {
      const float m2 = __shfl_xor(om, msk, 32);
      const float s2 = __shfl_xor(os, msk, 32);
      const float mn = fmaxf(om, m2);
      os = os * __expf(om - mn) + s2 * __expf(m2 - mn);
      om = mn;
    }
    rmax[r] = om;
    rinv[r] = 1.0f / os;
  }

  // ---------------- phase 2: recompute, emit P, accumulate P@V ------------
  v8f oacc[4] = {};
  for (int t = 0; t < nT; ++t) {
    const int c0 = t * 32;
    #pragma unroll
    for (int b = 0; b < 2; ++b) {
      const int col = c0 + 16 * b + ln;
      const int kf  = col / NPN;
      const __bf16* bp = k + col * DKN + 16 * g;
      v8f s = {};
      s = wmma_bf16(qa[0], cat16(*(const v8bf*)bp, *(const v8bf*)(bp + 8)), s);
      s = wmma_bf16(qa[1], cat16(*(const v8bf*)(bp + 32), *(const v8bf*)(bp + 40)), s);
      #pragma unroll
      for (int r = 0; r < 8; ++r) {
        const float p = (kf <= qf[r])
                          ? __expf(s[r] * 0.125f - rmax[r]) * rinv[r] : 0.f;
        at[(size_t)(m0 + 8 * g + r) * SN + col] = p;   // attention output (f32)
        P[8 * g + r][16 * b + ln] = (__bf16)p;          // staged for P@V
      }
    }

    // same-wave DS pipe is in-order; fence also stops compiler reordering
    asm volatile("s_wait_dscnt 0" ::: "memory");

    // A fragment of P (16x32 bf16) from LDS: two 16B ds reads
    const __bf16* pr = &P[ln][0];
    v16bf pa = cat16(*(const v8bf*)(pr + 16 * g), *(const v8bf*)(pr + 8 + 16 * g));

    // B fragments of V from the transposed copy: contiguous in key dim
    #pragma unroll
    for (int dt = 0; dt < 4; ++dt) {
      const __bf16* vp = vt + (dt * 16 + ln) * SN + c0 + 16 * g;
      v16bf vb = cat16(*(const v8bf*)vp, *(const v8bf*)(vp + 8));
      oacc[dt] = wmma_bf16(pa, vb, oacc[dt]);
    }

    asm volatile("s_wait_dscnt 0" ::: "memory");
  }

  // write concatenated head output [S, H*DK] (bf16) for the final projection
  #pragma unroll
  for (int dt = 0; dt < 4; ++dt) {
    #pragma unroll
    for (int r = 0; r < 8; ++r)
      oconb[(m0 + 8 * g + r) * DN + h * DKN + dt * 16 + ln] = (__bf16)oacc[dt][r];
  }

  // zero-fill fully masked attention columns (reference softmax yields 0)
  const int zs = nT * 32;
  const float4 z4 = {0.f, 0.f, 0.f, 0.f};
  for (int c = zs + g * 4; c < SN; c += 8)
    *(float4*)&at[(size_t)(m0 + ln) * SN + c] = z4;
}

// ---------------------------------------------------------------------------
extern "C" void kernel_launch(void* const* d_in, const int* in_sizes, int n_in,
                              void* d_out, int out_size, void* d_ws, size_t ws_size,
                              hipStream_t stream)
{
  const float* q  = (const float*)d_in[0];
  const float* kx = (const float*)d_in[1];
  const float* vx = (const float*)d_in[2];
  const float* Wq = (const float*)d_in[3];
  const float* bq = (const float*)d_in[4];
  const float* Wk = (const float*)d_in[5];
  const float* bk = (const float*)d_in[6];
  const float* Wv = (const float*)d_in[7];
  const float* bv = (const float*)d_in[8];
  const float* Wo = (const float*)d_in[9];
  const float* bo = (const float*)d_in[10];

  float* out = (float*)d_out;
  float* qh = out;
  float* kh = qh + (size_t)HN * SN * DKN;
  float* vh = kh + (size_t)HN * SN * DKN;
  float* at = vh + (size_t)HN * SN * DKN;
  float* fo = at + (size_t)HN * SN * SN;

  // bf16 workspace layout
  __bf16* w   = (__bf16*)d_ws;
  __bf16* xqb = w;              // [SN, DN] inputs
  __bf16* xkb = xqb + SD;
  __bf16* xvb = xkb + SD;
  __bf16* wqb = xvb + SD;       // [DN, DN] weights
  __bf16* wkb = wqb + WSZ;
  __bf16* wvb = wkb + WSZ;
  __bf16* wob = wvb + WSZ;
  __bf16* qhb = wob + WSZ;      // [H][SN][64] projected heads (bf16)
  __bf16* khb = qhb + SD;
  __bf16* vtb = khb + SD;       // [H][64][SN] V transposed (bf16)
  __bf16* ocb = vtb + SD;       // [SN, DN] concat head outputs

  const int gcS = (SD + 2047) / 2048;    // 8 elems/thread, 256 threads
  const int gcW = (WSZ + 2047) / 2048;
  cvt_f32_bf16<<<gcS, 256, 0, stream>>>(q,  xqb, SD);
  cvt_f32_bf16<<<gcS, 256, 0, stream>>>(kx, xkb, SD);
  cvt_f32_bf16<<<gcS, 256, 0, stream>>>(vx, xvb, SD);
  cvt_f32_bf16<<<gcW, 256, 0, stream>>>(Wq, wqb, WSZ);
  cvt_f32_bf16<<<gcW, 256, 0, stream>>>(Wk, wkb, WSZ);
  cvt_f32_bf16<<<gcW, 256, 0, stream>>>(Wv, wvb, WSZ);
  cvt_f32_bf16<<<gcW, 256, 0, stream>>>(Wo, wob, WSZ);

  const dim3 gg(SN / 16, DN / 16);       // 196 x 48 tiles
  gemm_bf16<0><<<gg, 32, 0, stream>>>(xqb, wqb, bq, qh, qhb);
  gemm_bf16<0><<<gg, 32, 0, stream>>>(xkb, wkb, bk, kh, khb);
  gemm_bf16<2><<<gg, 32, 0, stream>>>(xvb, wvb, bv, vh, vtb);

  attn_kernel<<<dim3(SN / 16, HN), 32, 0, stream>>>(qhb, khb, vtb, at, ocb);

  gemm_bf16<1><<<gg, 32, 0, stream>>>(ocb, wob, bo, fo, nullptr);
}
